// GAT_4layer_47218870453015
// MI455X (gfx1250) — compile-verified, matrix operand order checked by reading, not run on hardware
//
#include <hip/hip_runtime.h>
#include <hip/hip_bf16.h>
#include <cstdint>

typedef _Float16 v8h  __attribute__((ext_vector_type(8)));
typedef _Float16 v16h __attribute__((ext_vector_type(16)));
typedef float    v8f  __attribute__((ext_vector_type(8)));

#define NEG_SLOPE 0.2f
#define BN_EPS    1e-5f

// ---- monotone float<->uint encoding for atomic segment-max over signed floats
__device__ __forceinline__ unsigned f2o(float f) {
    unsigned u = __float_as_uint(f);
    return (u & 0x80000000u) ? ~u : (u | 0x80000000u);
}
__device__ __forceinline__ float o2f(unsigned o) {
    unsigned u = (o & 0x80000000u) ? (o & 0x7FFFFFFFu) : ~o;
    return __uint_as_float(u);
}

// ---------------------------------------------------------------- fills
__global__ void k_fill_u32(unsigned* __restrict__ p, unsigned v, long long n) {
    long long i = (long long)blockIdx.x * blockDim.x + threadIdx.x;
    if (i < n) p[i] = v;
}

// ---------------------------------------------------------------- f32 -> f16
__global__ void k_cvt_f16(const float* __restrict__ s, _Float16* __restrict__ d, long long n) {
    long long i = (long long)blockIdx.x * blockDim.x + threadIdx.x;
    if (i < n) d[i] = (_Float16)s[i];
}

// W[K x N] row-major  ->  Bt[Npad x K] f16 (row n holds column n of W; zero pad)
__global__ void k_cvt_wt(const float* __restrict__ W, _Float16* __restrict__ Bt,
                         int K, int N, int Npad) {
    int i = blockIdx.x * blockDim.x + threadIdx.x;
    if (i >= Npad * K) return;
    int n = i / K, k = i - n * K;
    Bt[i] = (n < N) ? (_Float16)W[(long long)k * N + n] : (_Float16)0.0f;
}

// ---------------------------------------------------------------- WMMA GEMM
// C[M x Nout] = A16[M x K] * Bt16[Npad x K]^T  (+bias) (+relu)
// one wave (32 lanes) per 16x16 tile; blockDim.x == 32
__global__ void k_gemm_wmma(const _Float16* __restrict__ A,
                            const _Float16* __restrict__ Bt,
                            float* __restrict__ C,
                            const float* __restrict__ bias,
                            int M, int K, int Nout, int relu) {
    const int lane = threadIdx.x;          // wave32
    const int r    = lane & 15;            // A row / B col within tile
    const int kb   = (lane >> 4) << 3;     // 0 or 8 (K sub-block per half-wave)
    const int mt   = blockIdx.x;
    const int nt   = blockIdx.y;
    const int ncol = nt * 16 + r;

    const _Float16* arow = A  + (long long)(mt * 16 + r) * K;
    const _Float16* brow = Bt + (long long)ncol * K;

    v8f acc = {};
    for (int kc = 0; kc < K; kc += 32) {
        v8h alo = *(const v8h*)(arow + kc + kb);
        v8h ahi = *(const v8h*)(arow + kc + 16 + kb);
        v8h blo = *(const v8h*)(brow + kc + kb);
        v8h bhi = *(const v8h*)(brow + kc + 16 + kb);
        v16h a = __builtin_shufflevector(alo, ahi, 0,1,2,3,4,5,6,7,8,9,10,11,12,13,14,15);
        v16h b = __builtin_shufflevector(blo, bhi, 0,1,2,3,4,5,6,7,8,9,10,11,12,13,14,15);
        acc = __builtin_amdgcn_wmma_f32_16x16x32_f16(
                  false, a, false, b, (short)0, acc, false, false);
    }

    if (ncol < Nout) {
        float bv = bias ? bias[ncol] : 0.0f;
        int mbase = mt * 16 + ((lane >> 4) << 3);
        #pragma unroll
        for (int i = 0; i < 8; ++i) {
            float v = acc[i] + bv;
            if (relu) v = v > 0.0f ? v : 0.0f;
            C[(long long)(mbase + i) * Nout + ncol] = v;
        }
    }
}

// ---------------------------------------------------------------- es/ed head dots
__global__ void k_esed(const float* __restrict__ h, const float* __restrict__ asrc,
                       const float* __restrict__ adst, float* __restrict__ es,
                       float* __restrict__ ed, int N) {
    int i = blockIdx.x * blockDim.x + threadIdx.x;
    if (i >= N * 4) return;
    int n = i >> 2, hh = i & 3;
    const float* hp = h + (long long)n * 256 + hh * 64;
    const float* ap = asrc + hh * 64;
    const float* bp = adst + hh * 64;
    float s = 0.0f, d = 0.0f;
    #pragma unroll 8
    for (int j = 0; j < 64; ++j) { float v = hp[j]; s += v * ap[j]; d += v * bp[j]; }
    es[i] = s; ed[i] = d;
}

// ---------------------------------------------------------------- edge pass 1: logits + segment max
__global__ void k_edge_logits(const float* __restrict__ es, const float* __restrict__ ed,
                              const long long* __restrict__ ei, long long E, int N,
                              float* __restrict__ ebuf, unsigned* __restrict__ mbuf) {
    long long e = (long long)blockIdx.x * blockDim.x + threadIdx.x;
    long long E2 = E + N;
    if (e >= E2) return;
    long long s, d;
    if (e < E) { s = ei[e]; d = ei[E + e]; } else { s = d = e - E; }
    #pragma unroll
    for (int hh = 0; hh < 4; ++hh) {
        float v = es[s * 4 + hh] + ed[d * 4 + hh];
        v = v > 0.0f ? v : NEG_SLOPE * v;
        ebuf[e * 4 + hh] = v;
        atomicMax(&mbuf[d * 4 + hh], f2o(v));
    }
}

// ---------------------------------------------------------------- edge pass 2: exp + segment sum
__global__ void k_edge_exp(float* __restrict__ ebuf, const unsigned* __restrict__ mbuf,
                           float* __restrict__ den, const long long* __restrict__ ei,
                           long long E, int N) {
    long long e = (long long)blockIdx.x * blockDim.x + threadIdx.x;
    long long E2 = E + N;
    if (e >= E2) return;
    long long d;
    if (e < E) { d = ei[E + e]; } else { d = e - E; }
    #pragma unroll
    for (int hh = 0; hh < 4; ++hh) {
        float ex = __expf(ebuf[e * 4 + hh] - o2f(mbuf[d * 4 + hh]));
        ebuf[e * 4 + hh] = ex;
        atomicAdd(&den[d * 4 + hh], ex);
    }
}

// ---------------------------------------------------------------- edge pass 3: alpha-weighted aggregation
// 32 sub-threads per edge, 8 contiguous channels each (8 | 64 so single head per sub)
__global__ void k_edge_aggr(const float* __restrict__ h, const float* __restrict__ ebuf,
                            const float* __restrict__ den, const long long* __restrict__ ei,
                            long long E, int N, float* __restrict__ aggr) {
    long long t = (long long)blockIdx.x * blockDim.x + threadIdx.x;
    long long E2 = E + N;
    if (t >= E2 * 32) return;
    long long e = t >> 5;
    int sub = (int)(t & 31);
    long long s, d;
    if (e < E) { s = ei[e]; d = ei[E + e]; } else { s = d = e - E; }
    int c0 = sub * 8;
    int hh = c0 >> 6;
    float alpha = ebuf[e * 4 + hh] / den[d * 4 + hh];
    const float4* hp = (const float4*)(h + (long long)s * 256 + c0);
    float4 x0 = hp[0], x1 = hp[1];
    float* op = aggr + (long long)d * 256 + c0;
    atomicAdd(op + 0, x0.x * alpha); atomicAdd(op + 1, x0.y * alpha);
    atomicAdd(op + 2, x0.z * alpha); atomicAdd(op + 3, x0.w * alpha);
    atomicAdd(op + 4, x1.x * alpha); atomicAdd(op + 5, x1.y * alpha);
    atomicAdd(op + 6, x1.z * alpha); atomicAdd(op + 7, x1.w * alpha);
}

// ---------------------------------------------------------------- BN stats (sum / sumsq per channel)
__global__ void k_bn_stats(const float* __restrict__ x, const float* __restrict__ bias,
                           float* __restrict__ stats, int N) {
    int c = threadIdx.x;          // blockDim.x == 256 channels
    float s = 0.0f, s2 = 0.0f;
    for (int n = blockIdx.x; n < N; n += gridDim.x) {
        float v = x[(long long)n * 256 + c] + bias[c];
        s += v; s2 += v * v;
    }
    atomicAdd(&stats[c], s);
    atomicAdd(&stats[256 + c], s2);
}

// ---------------------------------------------------------------- BN normalize + relu + residual accumulate
__global__ void k_bn_norm(const float* __restrict__ x, const float* __restrict__ bias,
                          const float* __restrict__ stats, const float* __restrict__ g,
                          const float* __restrict__ b, float* __restrict__ act,
                          float* __restrict__ z, int N) {
    long long i = (long long)blockIdx.x * blockDim.x + threadIdx.x;
    if (i >= (long long)N * 256) return;
    int c = (int)(i & 255);
    float mu  = stats[c] / (float)N;
    float var = stats[256 + c] / (float)N - mu * mu;
    float v = (x[i] + bias[c] - mu) * rsqrtf(var + BN_EPS) * g[c] + b[c];
    v = v > 0.0f ? v : 0.0f;
    act[i] = v;
    z[i] += v;
}

// ================================================================ host side
static inline unsigned gblk(long long n, int b) { return (unsigned)((n + b - 1) / b); }

extern "C" void kernel_launch(void* const* d_in, const int* in_sizes, int n_in,
                              void* d_out, int out_size, void* d_ws, size_t ws_size,
                              hipStream_t stream) {
    (void)in_sizes; (void)n_in; (void)out_size; (void)ws_size;
    const int  N  = 20000;
    const long long E = 320000;
    const long long E2 = E + N;

    const float*     x  = (const float*)d_in[0];
    const long long* ei = (const long long*)d_in[1];   // edge_index int64 [2,E]
    // d_in[2] = batch (unused)
    // per-layer params at 3 + 6*l: W, a_src, a_dst, b, bn_g, bn_b
    const float* lin1_w = (const float*)d_in[27];
    const float* lin1_b = (const float*)d_in[28];
    const float* lin2_w = (const float*)d_in[29];
    const float* lin2_b = (const float*)d_in[30];
    const float* lin3_w = (const float*)d_in[31];
    const float* lin3_b = (const float*)d_in[32];

    // ---- workspace layout (bytes, all 16B aligned)
    char* ws = (char*)d_ws;
    size_t o = 0;
    _Float16* A16 = (_Float16*)(ws + o); o += (size_t)N * 256 * 2;      // activations f16
    _Float16* W16 = (_Float16*)(ws + o); o += (size_t)1024 * 256 * 2;   // transposed weights f16
    float* H    = (float*)(ws + o); o += (size_t)N * 256 * 4;           // gemm out / mlp tmp
    float* AGGR = (float*)(ws + o); o += (size_t)N * 256 * 4;           // edge aggregation / mlp tmp2
    float* Z    = (float*)(ws + o); o += (size_t)N * 256 * 4;           // residual sum of layer outs
    float* ACT  = (float*)(ws + o); o += (size_t)N * 256 * 4;           // post-BN activation
    float* ES   = (float*)(ws + o); o += (size_t)N * 4 * 4;
    float* ED   = (float*)(ws + o); o += (size_t)N * 4 * 4;
    float* EB   = (float*)(ws + o); o += (size_t)E2 * 4 * 4;            // per-edge logits / exp
    unsigned* MB = (unsigned*)(ws + o); o += (size_t)N * 4 * 4;         // segment max (encoded)
    float* DEN  = (float*)(ws + o); o += (size_t)N * 4 * 4;
    float* ST   = (float*)(ws + o); o += (size_t)512 * 4;               // bn stats

    const unsigned NEG_INF_ENC = 0x007FFFFFu;                           // f2o(-inf)

    // zero residual accumulator
    k_fill_u32<<<gblk((long long)N * 256, 256), 256, 0, stream>>>((unsigned*)Z, 0u, (long long)N * 256);

    // layer-0 activation = x [N,128] -> f16
    k_cvt_f16<<<gblk((long long)N * 128, 256), 256, 0, stream>>>(x, A16, (long long)N * 128);

    for (int l = 0; l < 4; ++l) {
        const float* W    = (const float*)d_in[3 + 6 * l + 0];
        const float* asrc = (const float*)d_in[3 + 6 * l + 1];
        const float* adst = (const float*)d_in[3 + 6 * l + 2];
        const float* bb   = (const float*)d_in[3 + 6 * l + 3];
        const float* bng  = (const float*)d_in[3 + 6 * l + 4];
        const float* bnb  = (const float*)d_in[3 + 6 * l + 5];
        const int K = (l == 0) ? 128 : 256;

        // h = act @ W   (WMMA)
        k_cvt_wt<<<gblk(256 * K, 256), 256, 0, stream>>>(W, W16, K, 256, 256);
        {
            dim3 grid(N / 16, 256 / 16);
            k_gemm_wmma<<<grid, 32, 0, stream>>>(A16, W16, H, nullptr, N, K, 256, 0);
        }

        // attention logits per node/head
        k_esed<<<gblk((long long)N * 4, 256), 256, 0, stream>>>(H, asrc, adst, ES, ED, N);

        // edge softmax + aggregation
        k_fill_u32<<<gblk((long long)N * 4, 256), 256, 0, stream>>>(MB, NEG_INF_ENC, (long long)N * 4);
        k_fill_u32<<<gblk((long long)N * 4, 256), 256, 0, stream>>>((unsigned*)DEN, 0u, (long long)N * 4);
        k_edge_logits<<<gblk(E2, 256), 256, 0, stream>>>(ES, ED, ei, E, N, EB, MB);
        k_edge_exp<<<gblk(E2, 256), 256, 0, stream>>>(EB, MB, DEN, ei, E, N);
        k_fill_u32<<<gblk((long long)N * 256, 256), 256, 0, stream>>>((unsigned*)AGGR, 0u, (long long)N * 256);
        k_edge_aggr<<<gblk(E2 * 32, 256), 256, 0, stream>>>(H, EB, DEN, ei, E, N, AGGR);

        // BN + relu + residual accumulate; next-layer activation
        k_fill_u32<<<gblk(512, 256), 256, 0, stream>>>((unsigned*)ST, 0u, 512);
        k_bn_stats<<<256, 256, 0, stream>>>(AGGR, bb, ST, N);
        k_bn_norm<<<gblk((long long)N * 256, 256), 256, 0, stream>>>(AGGR, bb, ST, bng, bnb, ACT, Z, N);
        k_cvt_f16<<<gblk((long long)N * 256, 256), 256, 0, stream>>>(ACT, A16, (long long)N * 256);
    }

    // ---- MLP head
    // lin1: relu(z @ W1 + b1) -> H
    k_cvt_f16<<<gblk((long long)N * 256, 256), 256, 0, stream>>>(Z, A16, (long long)N * 256);
    k_cvt_wt<<<gblk(256 * 256, 256), 256, 0, stream>>>(lin1_w, W16, 256, 256, 256);
    { dim3 g(N / 16, 16); k_gemm_wmma<<<g, 32, 0, stream>>>(A16, W16, H, lin1_b, N, 256, 256, 1); }

    // lin2: relu(H @ W2 + b2) -> AGGR [N,128]
    k_cvt_f16<<<gblk((long long)N * 256, 256), 256, 0, stream>>>(H, A16, (long long)N * 256);
    k_cvt_wt<<<gblk(128 * 256, 256), 256, 0, stream>>>(lin2_w, W16, 256, 128, 128);
    { dim3 g(N / 16, 8); k_gemm_wmma<<<g, 32, 0, stream>>>(A16, W16, AGGR, lin2_b, N, 256, 128, 1); }

    // lin3: AGGR @ W3 + b3 -> d_out [N,978]  (pad N to 992)
    k_cvt_f16<<<gblk((long long)N * 128, 256), 256, 0, stream>>>(AGGR, A16, (long long)N * 128);
    k_cvt_wt<<<gblk(992 * 128, 256), 256, 0, stream>>>(lin3_w, W16, 128, 978, 992);
    { dim3 g(N / 16, 992 / 16); k_gemm_wmma<<<g, 32, 0, stream>>>(A16, W16, (float*)d_out, lin3_b, N, 128, 978, 0); }
}
